// Mistral4Attention_67422396613251
// MI455X (gfx1250) — matrix operand order repaired and checked by reading, not compiled, not run on hardware
//
#include <hip/hip_runtime.h>
#include <hip/hip_bf16.h>
#include <math.h>

// ---------------------------------------------------------------------------
// MLA attention block for MI455X (gfx1250)
//  - bf16 WMMA (v_wmma_f32_16x16x32_bf16) for all matmuls
//  - gfx1250 async-to-LDS copies (global_load_async_to_lds_b128 / ASYNCcnt)
//    with double-buffered GEMM mainloop; loop-invariant addresses hoisted
// ---------------------------------------------------------------------------

typedef __attribute__((ext_vector_type(16))) __bf16 v16bf;
typedef __attribute__((ext_vector_type(4)))  __bf16 v4bf;
typedef __attribute__((ext_vector_type(8)))  float  v8f;

#define S_LEN 2048
#define H_HEADS 32

// ---------------------------------------------------------------------------
// gfx1250 async copy: HBM -> LDS, 16B per lane, tracked by ASYNCcnt.
// LDS generic pointers carry the LDS offset in their low 32 bits.
// ---------------------------------------------------------------------------
__device__ __forceinline__ void async_copy_b128(const __bf16* g, __bf16* l) {
  unsigned lds = (unsigned)(unsigned long long)l;
  asm volatile("global_load_async_to_lds_b128 %0, %1, off"
               :: "v"(lds), "v"(g) : "memory");
}
__device__ __forceinline__ void wait_async0() {
  asm volatile("s_wait_asynccnt 0x0" ::: "memory");
}
__device__ __forceinline__ void wait_async4() {
  asm volatile("s_wait_asynccnt 0x4" ::: "memory");
}

// ---------------------------------------------------------------------------
// CDNA5 WMMA fragment loaders (wave32); A and B both use contiguous k = 16*(lane/16) + i.
// ---------------------------------------------------------------------------
__device__ __forceinline__ v16bf frag_a_rowmajor(const __bf16* lds, int ld,
                                                 int row0, int k0, int lane) {
  const __bf16* p = lds + (row0 + (lane & 15)) * ld;
  int kb = k0 + ((lane < 16) ? 0 : 16);
  v16bf a;
#pragma unroll
  for (int i = 0; i < 16; ++i) a[i] = p[kb + i];
  return a;
}

// B operand stored as (N rows x K cols) row-major: column n of B == stored row n.
__device__ __forceinline__ v16bf frag_b_nk(const __bf16* lds, int ld,
                                           int n0, int k0, int lane) {
  const __bf16* p = lds + (n0 + (lane & 15)) * ld + k0 + ((lane < 16) ? 0 : 16);
  v16bf b;
#pragma unroll
  for (int i = 0; i < 16; ++i) b[i] = p[i];
  return b;
}

// B operand stored as (K rows x N cols) row-major (the V tile).
__device__ __forceinline__ v16bf frag_b_kn(const __bf16* lds, int ld,
                                           int n0, int k0, int lane) {
  int n  = n0 + (lane & 15);
  int kb = k0 + ((lane < 16) ? 0 : 16);
  v16bf b;
#pragma unroll
  for (int i = 0; i < 16; ++i) b[i] = lds[(kb + i) * ld + n];
  return b;
}

__device__ __forceinline__ v8f wmma_bf16(v16bf a, v16bf b, v8f c) {
  return __builtin_amdgcn_wmma_f32_16x16x32_bf16(false, a, false, b,
                                                 (short)0, c, false, false);
}

// ---------------------------------------------------------------------------
// f32 -> bf16 conversion (one shot per tensor; halves HBM traffic for GEMMs)
// ---------------------------------------------------------------------------
__global__ void __launch_bounds__(256)
cvt_f32_bf16_kernel(const float* __restrict__ in, __bf16* __restrict__ out,
                    int n4) {
  int i = blockIdx.x * 256 + threadIdx.x;
  if (i < n4) {
    float4 v = ((const float4*)in)[i];
    v4bf o;
    o[0] = (__bf16)v.x; o[1] = (__bf16)v.y;
    o[2] = (__bf16)v.z; o[3] = (__bf16)v.w;
    ((v4bf*)out)[i] = o;
  }
}

// ---------------------------------------------------------------------------
// GEMM: C[M,N] = A[M,K] * B[N,K]^T, A/B bf16, C f32.
// 256 threads = 8 waves (4 M x 2 N), wave tile 32x64, block tile 128x128x32.
// Double-buffered async-to-LDS mainloop (prefetch next K-tile during WMMAs).
// Per-thread copy addresses are k-loop invariant and hoisted.
// M divides 128 and K divides 32 for every call here; N guarded.
// ---------------------------------------------------------------------------
#define GBM 128
#define GBN 128
#define GBK 32
#define GLD 40  // GBK + 8 bf16 pad; 80B rows keep 16B-aligned b128 targets

__global__ void __launch_bounds__(256)
gemm_nt_bf16_async(const __bf16* __restrict__ A, const __bf16* __restrict__ B,
                   float* __restrict__ C, int M, int N, int K) {
  __shared__ alignas(16) __bf16 Ash[2][GBM * GLD];
  __shared__ alignas(16) __bf16 Bsh[2][GBN * GLD];

  const int tid  = threadIdx.x;
  const int lane = tid & 31;
  const int wave = tid >> 5;
  const int wm   = (wave & 3) * 32;
  const int wn   = (wave >> 2) * 64;
  const int m0   = blockIdx.y * GBM;
  const int n0   = blockIdx.x * GBN;

  // Per-thread async-copy coordinates (2 b128 chunks per matrix per tile).
  const int r0 = tid >> 2;            // rows 0..63
  const int r1 = r0 + 64;             // rows 64..127
  const int c0 = (tid & 3) * 8;       // chunk column (bf16 elems)
  const int rb0 = (n0 + r0 < N) ? r0 : 0;  // clamp; cols masked at store
  const int rb1 = (n0 + r1 < N) ? r1 : 0;
  const __bf16* gA0 = A + (size_t)(m0 + r0) * K + c0;
  const __bf16* gA1 = A + (size_t)(m0 + r1) * K + c0;
  const __bf16* gB0 = B + (size_t)(n0 + rb0) * K + c0;
  const __bf16* gB1 = B + (size_t)(n0 + rb1) * K + c0;
  const int lA0 = r0 * GLD + c0;
  const int lA1 = r1 * GLD + c0;

  v8f acc[2][4];
  v8f zero = {0.f, 0.f, 0.f, 0.f, 0.f, 0.f, 0.f, 0.f};
#pragma unroll
  for (int i = 0; i < 2; ++i)
#pragma unroll
    for (int j = 0; j < 4; ++j) acc[i][j] = zero;

  auto issue = [&](int buf, int kt) {
    const int k0 = kt * GBK;
    async_copy_b128(gA0 + k0, &Ash[buf][lA0]);
    async_copy_b128(gA1 + k0, &Ash[buf][lA1]);
    async_copy_b128(gB0 + k0, &Bsh[buf][lA0]);
    async_copy_b128(gB1 + k0, &Bsh[buf][lA1]);
  };

  const int nk = K / GBK;
  issue(0, 0);
  int cur = 0;
  for (int kt = 0; kt < nk; ++kt) {
    if (kt + 1 < nk) {
      issue(cur ^ 1, kt + 1);
      wait_async4();  // oldest 4 (current tile) done; prefetch stays in flight
    } else {
      wait_async0();
    }
    __syncthreads();

    const __bf16* As = Ash[cur];
    const __bf16* Bs = Bsh[cur];
    v16bf a0 = frag_a_rowmajor(As, GLD, wm,      0, lane);
    v16bf a1 = frag_a_rowmajor(As, GLD, wm + 16, 0, lane);
    v16bf b0 = frag_b_nk(Bs, GLD, wn,      0, lane);
    v16bf b1 = frag_b_nk(Bs, GLD, wn + 16, 0, lane);
    v16bf b2 = frag_b_nk(Bs, GLD, wn + 32, 0, lane);
    v16bf b3 = frag_b_nk(Bs, GLD, wn + 48, 0, lane);
    acc[0][0] = wmma_bf16(a0, b0, acc[0][0]);
    acc[0][1] = wmma_bf16(a0, b1, acc[0][1]);
    acc[0][2] = wmma_bf16(a0, b2, acc[0][2]);
    acc[0][3] = wmma_bf16(a0, b3, acc[0][3]);
    acc[1][0] = wmma_bf16(a1, b0, acc[1][0]);
    acc[1][1] = wmma_bf16(a1, b1, acc[1][1]);
    acc[1][2] = wmma_bf16(a1, b2, acc[1][2]);
    acc[1][3] = wmma_bf16(a1, b3, acc[1][3]);
    __syncthreads();
    cur ^= 1;
  }

  const int rbase = (lane < 16) ? 0 : 8;
  const int ncol  = lane & 15;
#pragma unroll
  for (int i = 0; i < 2; ++i)
#pragma unroll
    for (int j = 0; j < 4; ++j) {
      int col = n0 + wn + j * 16 + ncol;
      if (col < N) {
#pragma unroll
        for (int r = 0; r < 8; ++r)
          C[(size_t)(m0 + wm + i * 16 + rbase + r) * N + col] = acc[i][j][r];
      }
    }
}

// ---------------------------------------------------------------------------
// RMSNorm over rows: out[i] = in[i] * rsqrt(mean(in^2) + 1e-6) * w[i]
// ---------------------------------------------------------------------------
template <typename OutT>
__global__ void __launch_bounds__(256)
rmsnorm_kernel(const float* __restrict__ in, const float* __restrict__ w,
               OutT* __restrict__ out, int L, int in_stride, int out_stride) {
  const int row = blockIdx.x;
  const float* xr = in + (size_t)row * in_stride;
  float ss = 0.f;
  for (int i = threadIdx.x; i < L; i += 256) {
    float v = xr[i];
    ss += v * v;
  }
#pragma unroll
  for (int off = 16; off > 0; off >>= 1) ss += __shfl_xor(ss, off, 32);
  __shared__ float red[8];
  int lane = threadIdx.x & 31, wv = threadIdx.x >> 5;
  if (lane == 0) red[wv] = ss;
  __syncthreads();
  float tot = 0.f;
#pragma unroll
  for (int i = 0; i < 8; ++i) tot += red[i];
  float inv = rsqrtf(tot / (float)L + 1e-6f);
  OutT* orow = out + (size_t)row * out_stride;
  for (int i = threadIdx.x; i < L; i += 256) orow[i] = (OutT)(xr[i] * inv * w[i]);
}

// ---------------------------------------------------------------------------
// YaRN RoPE + pack into per-head-contiguous bf16 tensors (H, S, 128).
// ---------------------------------------------------------------------------
__global__ void __launch_bounds__(128)
rope_pack_kernel(const float* __restrict__ q, const float* __restrict__ kv,
                 const float* __restrict__ ckv_kpe,
                 __bf16* __restrict__ Qb, __bf16* __restrict__ Kb,
                 __bf16* __restrict__ Vb) {
  const int s   = blockIdx.x;
  const int tid = threadIdx.x;
  __shared__ float cbuf[32], sbuf[32];

  if (tid < 32) {
    const float j = (float)tid;
    const float PI2 = 6.2831853071795864769f;
    const float denom = 2.f * logf(10000.f);
    float low  = floorf(64.f * logf(8192.f / (32.f * PI2)) / denom);
    float high = ceilf(64.f * logf(8192.f / (1.f * PI2)) / denom);
    low  = fmaxf(low, 0.f);
    high = fminf(high, 63.f);
    float hi = (low == high) ? high + 0.001f : high;
    float ramp = fminf(fmaxf((j - low) / (hi - low), 0.f), 1.f);
    float extra = powf(10000.f, -j / 32.f);
    float inv   = (extra / 128.f) * ramp + extra * (1.f - ramp);  // mscale == 1
    float ang = (float)s * inv;
    cbuf[tid] = cosf(ang);
    sbuf[tid] = sinf(ang);
  }
  __syncthreads();

  const float* qrow = q + (size_t)s * (H_HEADS * 128);
  const float* kvro = kv + (size_t)s * (H_HEADS * 192);
  const float* kpe  = ckv_kpe + (size_t)s * 320 + 256;

  for (int idx = tid; idx < H_HEADS * 128; idx += 128) {
    int h = idx >> 7, d = idx & 127;
    size_t o = ((size_t)h * S_LEN + s) * 128 + d;

    float qv;
    if (d < 64) {
      qv = qrow[h * 128 + d];
    } else {
      const float* qp = qrow + h * 128 + 64;
      int i = d - 64;
      if (i < 32) qv = qp[2 * i] * cbuf[i] - qp[2 * i + 1] * sbuf[i];
      else { i -= 32; qv = qp[2 * i + 1] * cbuf[i] + qp[2 * i] * sbuf[i]; }
    }
    Qb[o] = (__bf16)qv;

    float kvv;
    if (d < 64) {
      kvv = kvro[h * 192 + d];
    } else {
      int i = d - 64;
      if (i < 32) kvv = kpe[2 * i] * cbuf[i] - kpe[2 * i + 1] * sbuf[i];
      else { i -= 32; kvv = kpe[2 * i + 1] * cbuf[i] + kpe[2 * i] * sbuf[i]; }
    }
    Kb[o] = (__bf16)kvv;

    Vb[o] = (__bf16)kvro[h * 192 + 64 + d];
  }
}

// ---------------------------------------------------------------------------
// Flash attention (causal, online softmax). One block per (64-query tile, head).
// Tiles staged with async-to-LDS; O emitted as bf16 for the output GEMM.
// ---------------------------------------------------------------------------
#define FLD 136  // 128 + 8 bf16 pad (272B rows: 16B multiple)
#define PLD 72   // 64 + 8 bf16 pad

__global__ void __launch_bounds__(128)
mla_flash_kernel(const __bf16* __restrict__ Qb, const __bf16* __restrict__ Kb,
                 const __bf16* __restrict__ Vb, __bf16* __restrict__ O,
                 float scale) {
  __shared__ alignas(16) __bf16 Qs[64 * FLD];
  __shared__ alignas(16) __bf16 Ks[64 * FLD];
  __shared__ alignas(16) __bf16 Vs[64 * FLD];
  __shared__ alignas(16) __bf16 Ps[4 * 16 * PLD];

  const int tid = threadIdx.x, lane = tid & 31, wave = tid >> 5;
  const int h  = blockIdx.y;
  const int q0 = blockIdx.x * 64;
  const int qw = q0 + wave * 16;
  const int half8 = (lane < 16) ? 0 : 8;
  const int l15   = lane & 15;

  // Per-thread tile-copy coordinates (4 b128 chunks per 64x128 tile).
  const int fr = tid >> 3;            // rows 0..15 (stride 16 over 4 chunks)
  const int fc = (tid & 7) * 16;      // chunk column (bf16 elems)

  // Q tile
  const __bf16* Qg = Qb + ((size_t)h * S_LEN + q0) * 128 + (size_t)fr * 128 + fc;
#pragma unroll
  for (int it = 0; it < 4; ++it)
    async_copy_b128(Qg + (size_t)it * (16 * 128), &Qs[(fr + 16 * it) * FLD + fc]);
  wait_async0();
  __syncthreads();

  v8f zero = {0.f, 0.f, 0.f, 0.f, 0.f, 0.f, 0.f, 0.f};
  v8f o[8];
#pragma unroll
  for (int n = 0; n < 8; ++n) o[n] = zero;
  float mrow[8], lrow[8];
#pragma unroll
  for (int r = 0; r < 8; ++r) { mrow[r] = -3.0e38f; lrow[r] = 0.f; }

  // Hoisted per-thread K/V copy base pointers (advance by 64 rows per tile).
  const __bf16* Kg = Kb + (size_t)h * S_LEN * 128 + (size_t)fr * 128 + fc;
  const __bf16* Vg = Vb + (size_t)h * S_LEN * 128 + (size_t)fr * 128 + fc;

  const int nkt = blockIdx.x + 1;  // causal: key tiles 0..q-tile
  for (int kt = 0; kt < nkt; ++kt) {
    const int kbase = kt * 64;
    const __bf16* Kt = Kg + (size_t)kbase * 128;
    const __bf16* Vt = Vg + (size_t)kbase * 128;
#pragma unroll
    for (int it = 0; it < 4; ++it) {
      async_copy_b128(Kt + (size_t)it * (16 * 128), &Ks[(fr + 16 * it) * FLD + fc]);
      async_copy_b128(Vt + (size_t)it * (16 * 128), &Vs[(fr + 16 * it) * FLD + fc]);
    }
    wait_async0();
    __syncthreads();

    // S = Q(16x128) * K^T -> 4 fragments of 16x16
    v8f sc[4];
#pragma unroll
    for (int n = 0; n < 4; ++n) sc[n] = zero;
#pragma unroll
    for (int kk = 0; kk < 4; ++kk) {
      v16bf a = frag_a_rowmajor(Qs, FLD, wave * 16, kk * 32, lane);
#pragma unroll
      for (int n = 0; n < 4; ++n) {
        v16bf b = frag_b_nk(Ks, FLD, n * 16, kk * 32, lane);
        sc[n] = wmma_bf16(a, b, sc[n]);
      }
    }

    // scale + causal mask
#pragma unroll
    for (int n = 0; n < 4; ++n)
#pragma unroll
      for (int r = 0; r < 8; ++r) {
        float v = sc[n][r] * scale;
        int qrow = qw + half8 + r;
        int kcol = kbase + n * 16 + l15;
        if (kcol > qrow) v = -3.0e38f;
        sc[n][r] = v;
      }

    // row max over the 16-lane half sharing each row
    float mnew[8];
#pragma unroll
    for (int r = 0; r < 8; ++r) {
      float mx = fmaxf(fmaxf(sc[0][r], sc[1][r]), fmaxf(sc[2][r], sc[3][r]));
      mx = fmaxf(mx, __shfl_xor(mx, 1, 32));
      mx = fmaxf(mx, __shfl_xor(mx, 2, 32));
      mx = fmaxf(mx, __shfl_xor(mx, 4, 32));
      mx = fmaxf(mx, __shfl_xor(mx, 8, 32));
      mnew[r] = fmaxf(mrow[r], mx);
    }

    // rescale running state
#pragma unroll
    for (int r = 0; r < 8; ++r) {
      float alpha = expf(mrow[r] - mnew[r]);
      lrow[r] *= alpha;
#pragma unroll
      for (int n = 0; n < 8; ++n) o[n][r] *= alpha;
      mrow[r] = mnew[r];
    }

    // p = exp(s - m); stage P per-wave in LDS (C-layout -> A-layout transpose)
    float rsum[8];
#pragma unroll
    for (int r = 0; r < 8; ++r) rsum[r] = 0.f;
    __bf16* Pw = Ps + wave * 16 * PLD;
#pragma unroll
    for (int n = 0; n < 4; ++n)
#pragma unroll
      for (int r = 0; r < 8; ++r) {
        float p = expf(sc[n][r] - mrow[r]);
        rsum[r] += p;
        Pw[(half8 + r) * PLD + n * 16 + l15] = (__bf16)p;
      }
#pragma unroll
    for (int r = 0; r < 8; ++r) {
      float s = rsum[r];
      s += __shfl_xor(s, 1, 32);
      s += __shfl_xor(s, 2, 32);
      s += __shfl_xor(s, 4, 32);
      s += __shfl_xor(s, 8, 32);
      lrow[r] += s;
    }
    __syncthreads();

    // O += P(16x64) * V(64x128)
#pragma unroll
    for (int kk = 0; kk < 2; ++kk) {
      v16bf a = frag_a_rowmajor(Pw, PLD, 0, kk * 32, lane);
#pragma unroll
      for (int n = 0; n < 8; ++n) {
        v16bf b = frag_b_kn(Vs, FLD, n * 16, kk * 32, lane);
        o[n] = wmma_bf16(a, b, o[n]);
      }
    }
    __syncthreads();
  }

  // normalize, store O as (S, H*128) bf16
  __bf16* Og = O + (size_t)qw * (H_HEADS * 128) + h * 128;
#pragma unroll
  for (int n = 0; n < 8; ++n)
#pragma unroll
    for (int r = 0; r < 8; ++r)
      Og[(size_t)(half8 + r) * (H_HEADS * 128) + n * 16 + l15] =
          (__bf16)(o[n][r] / lrow[r]);
}

// ---------------------------------------------------------------------------
// Host launch
// ---------------------------------------------------------------------------
extern "C" void kernel_launch(void* const* d_in, const int* in_sizes, int n_in,
                              void* d_out, int out_size, void* d_ws, size_t ws_size,
                              hipStream_t stream) {
  (void)in_sizes; (void)n_in; (void)out_size; (void)ws_size;

  const float* x        = (const float*)d_in[0];  // (2048, 4096)
  const float* wq_a     = (const float*)d_in[1];  // (1024, 4096)
  const float* q_a_ln_w = (const float*)d_in[2];  // (1024,)
  const float* wq_b     = (const float*)d_in[3];  // (4096, 1024)
  const float* wkv_a    = (const float*)d_in[4];  // (320, 4096)
  const float* kv_a_ln  = (const float*)d_in[5];  // (256,)
  const float* wkv_b    = (const float*)d_in[6];  // (6144, 256)
  const float* wo       = (const float*)d_in[7];  // (4096, 4096)
  float* out            = (float*)d_out;          // (2048, 4096)

  char* wp = (char*)d_ws;
  auto carve = [&wp](size_t bytes) -> char* {
    char* p = wp;
    wp += (bytes + 255) & ~(size_t)255;
    return p;
  };
  __bf16* xb    = (__bf16*)carve((size_t)S_LEN * 4096 * 2);
  __bf16* wqab  = (__bf16*)carve((size_t)1024 * 4096 * 2);
  __bf16* wqbb  = (__bf16*)carve((size_t)4096 * 1024 * 2);
  __bf16* wkvab = (__bf16*)carve((size_t)320 * 4096 * 2);
  __bf16* wkvbb = (__bf16*)carve((size_t)6144 * 256 * 2);
  __bf16* wob   = (__bf16*)carve((size_t)4096 * 4096 * 2);
  float*  q_a   = (float*)carve((size_t)S_LEN * 1024 * 4);
  __bf16* qnb   = (__bf16*)carve((size_t)S_LEN * 1024 * 2);
  float*  q     = (float*)carve((size_t)S_LEN * 4096 * 4);
  float*  ckv   = (float*)carve((size_t)S_LEN * 320 * 4);
  __bf16* ckvnb = (__bf16*)carve((size_t)S_LEN * 256 * 2);
  float*  kvb   = (float*)carve((size_t)S_LEN * 6144 * 4);
  __bf16* Qb    = (__bf16*)carve((size_t)H_HEADS * S_LEN * 128 * 2);
  __bf16* Kb    = (__bf16*)carve((size_t)H_HEADS * S_LEN * 128 * 2);
  __bf16* Vb    = (__bf16*)carve((size_t)H_HEADS * S_LEN * 128 * 2);
  __bf16* Obuf  = (__bf16*)carve((size_t)S_LEN * 4096 * 2);

  const float m = 0.1f * logf(128.f) + 1.f;
  const float softmax_scale = (1.f / sqrtf(128.f)) * m * m;

  auto cvt = [&](const float* src, __bf16* dst, size_t n) {
    int n4 = (int)(n / 4);
    cvt_f32_bf16_kernel<<<(n4 + 255) / 256, 256, 0, stream>>>(src, dst, n4);
  };
  cvt(x,     xb,    (size_t)S_LEN * 4096);
  cvt(wq_a,  wqab,  (size_t)1024 * 4096);
  cvt(wq_b,  wqbb,  (size_t)4096 * 1024);
  cvt(wkv_a, wkvab, (size_t)320 * 4096);
  cvt(wkv_b, wkvbb, (size_t)6144 * 256);
  cvt(wo,    wob,   (size_t)4096 * 4096);

  // q_a = x @ wq_a^T
  gemm_nt_bf16_async<<<dim3(1024 / GBN, S_LEN / GBM), 256, 0, stream>>>(
      xb, wqab, q_a, S_LEN, 1024, 4096);
  // qnb = rmsnorm(q_a) (bf16 for next GEMM)
  rmsnorm_kernel<__bf16><<<S_LEN, 256, 0, stream>>>(q_a, q_a_ln_w, qnb,
                                                    1024, 1024, 1024);
  // q = qnb @ wq_b^T
  gemm_nt_bf16_async<<<dim3(4096 / GBN, S_LEN / GBM), 256, 0, stream>>>(
      qnb, wqbb, q, S_LEN, 4096, 1024);
  // ckv_kpe = x @ wkv_a^T  (N=320, guarded)
  gemm_nt_bf16_async<<<dim3((320 + GBN - 1) / GBN, S_LEN / GBM), 256, 0, stream>>>(
      xb, wkvab, ckv, S_LEN, 320, 4096);
  // ckvnb = rmsnorm(ckv[:, :256])
  rmsnorm_kernel<__bf16><<<S_LEN, 256, 0, stream>>>(ckv, kv_a_ln, ckvnb,
                                                    256, 320, 256);
  // kv = ckvnb @ wkv_b^T
  gemm_nt_bf16_async<<<dim3(6144 / GBN, S_LEN / GBM), 256, 0, stream>>>(
      ckvnb, wkvbb, kvb, S_LEN, 6144, 256);
  // RoPE + pack Q/K/V to (H, S, 128) bf16
  rope_pack_kernel<<<S_LEN, 128, 0, stream>>>(q, kvb, ckv, Qb, Kb, Vb);
  // causal flash attention -> Obuf (S, H*128) bf16
  mla_flash_kernel<<<dim3(S_LEN / 64, H_HEADS), 128, 0, stream>>>(
      Qb, Kb, Vb, Obuf, softmax_scale);
  // out = Obuf @ wo^T
  gemm_nt_bf16_async<<<dim3(4096 / GBN, S_LEN / GBM), 256, 0, stream>>>(
      Obuf, wob, out, S_LEN, 4096, 4096);
}
